// LinearAttention_82721070121149
// MI455X (gfx1250) — matrix operand order repaired and checked by reading, not compiled, not run on hardware
//
#include <hip/hip_runtime.h>
#include <stdint.h>

// Problem constants (fixed by the reference)
#define BB 32
#define CC 512
#define SS 1024
#define NH 8
#define HD 64

typedef __attribute__((ext_vector_type(16))) __bf16 v16bf;
typedef __attribute__((ext_vector_type(8)))  float  v8f;

__device__ __forceinline__ uint16_t f2bf(float f) {
    uint32_t u = __float_as_uint(f);
    uint32_t r = u + 0x7FFFu + ((u >> 16) & 1u);   // RNE
    return (uint16_t)(r >> 16);
}
__device__ __forceinline__ float bf2f(uint16_t u) {
    return __uint_as_float(((uint32_t)u) << 16);
}
__device__ __forceinline__ float phi(float v) {   // elu(v)+1
    return v > 0.f ? v + 1.f : __expf(v);
}

// ---- gfx1250 async global->LDS copy (ASYNCcnt path), with safe fallback ----
#if __has_builtin(__builtin_amdgcn_global_load_async_to_lds_b128)
#define ASYNC_LDS 1
#else
#define ASYNC_LDS 0
#endif

typedef int i4_vs __attribute__((vector_size(16)));
typedef __attribute__((address_space(1))) i4_vs* g_i4p;   // global int4*
typedef __attribute__((address_space(3))) i4_vs* l_i4p;   // LDS int4*

__device__ __forceinline__ void cp16_g2l(const void* g, void* l) {
#if ASYNC_LDS
    __builtin_amdgcn_global_load_async_to_lds_b128(
        (g_i4p)(uintptr_t)g,   // generic->AS(1): numeric identity
        (l_i4p)l,              // generic->AS(3): proper addrspacecast
        0, 0);
#else
    *(float4*)l = *(const float4*)g;
#endif
}
__device__ __forceinline__ void wait_async_lds() {
#if ASYNC_LDS
#if __has_builtin(__builtin_amdgcn_s_wait_asynccnt)
    __builtin_amdgcn_s_wait_asynccnt(0);
#else
    asm volatile("s_wait_asynccnt 0" ::: "memory");
#endif
#endif
}

// ---------------- helper kernels ----------------

__global__ void k_cvt4(const float* __restrict__ a, const float* __restrict__ b,
                       const float* __restrict__ c, const float* __restrict__ d,
                       uint16_t* __restrict__ oa, uint16_t* __restrict__ ob,
                       uint16_t* __restrict__ oc, uint16_t* __restrict__ od, int n) {
    int i = blockIdx.x * blockDim.x + threadIdx.x;
    if (i < n) { oa[i] = f2bf(a[i]); ob[i] = f2bf(b[i]); oc[i] = f2bf(c[i]); od[i] = f2bf(d[i]); }
}

// t[b][s][c] = bf16(x[b][c][s])   (LDS-tiled 32x32 transpose)
__global__ void k_transpose(const float* __restrict__ x, uint16_t* __restrict__ t) {
    __shared__ float tile[32][33];
    const int b = blockIdx.z;
    const int s0 = blockIdx.x * 32, c0 = blockIdx.y * 32;
    const int tx = threadIdx.x, ty = threadIdx.y;   // block (32, 8)
    #pragma unroll
    for (int i = 0; i < 32; i += 8)
        tile[ty + i][tx] = x[((size_t)b * CC + (c0 + ty + i)) * SS + s0 + tx];
    __syncthreads();
    #pragma unroll
    for (int i = 0; i < 32; i += 8)
        t[((size_t)b * SS + (s0 + ty + i)) * CC + c0 + tx] = f2bf(tile[tx][ty + i]);
}

// ksum[z][d] = sum_s Ktphi[z][d][s]  (one wave per row of 1024 bf16)
__global__ void k_ksum(const uint16_t* __restrict__ Kt, float* __restrict__ ksum) {
    const int row  = blockIdx.x * (blockDim.x >> 5) + (threadIdx.x >> 5);
    const int lane = threadIdx.x & 31;
    const uint16_t* p = Kt + (size_t)row * SS;
    float sum = 0.f;
    for (int i = lane; i < SS; i += 32) sum += bf2f(p[i]);
    #pragma unroll
    for (int off = 16; off; off >>= 1) sum += __shfl_xor(sum, off, 32);
    if (lane == 0) ksum[row] = sum;
}

// den[z][s] = dot(Qphi[z][s][:], ksum[z][:])
__global__ void k_den(const uint16_t* __restrict__ Q, const float* __restrict__ ksum,
                      float* __restrict__ den) {
    const int idx = blockIdx.x * blockDim.x + threadIdx.x;   // z*S + s
    const int z   = idx >> 10;
    const uint16_t* q = Q + (size_t)idx * HD;
    const float* ks   = ksum + (size_t)z * HD;
    float sum = 0.f;
    #pragma unroll
    for (int d = 0; d < HD; ++d) sum += bf2f(q[d]) * ks[d];
    den[idx] = sum;
}

// ---------------- shared GEMM pieces ----------------
// NT-GEMM: D[M][N] = sum_k A[m][k] * Bt[n][k]  (both operands row-major along K)
// Fragment layouts used here (contiguous k, same order for A and B):
//   A 16x32 bf16: lane g*16+m -> A[m][k0+16g..+15]
//   B 32x16 bf16: lane g*16+n -> Bt[n][k0+16g..+15]
//   C/D 16x16 f32: VGPR r, lane g*16+n -> D[8g+r][n]

enum { EPI_Q = 0, EPI_K, EPI_V, EPI_KVT, EPI_NUM, EPI_OUT };

union ABFrag { v16bf v; float4 q[2]; };
union Pack8  { uint16_t h[8]; uint4 u; };

__device__ __forceinline__ v8f wmma_bf16(const ABFrag& a, const ABFrag& b, v8f c) {
    return __builtin_amdgcn_wmma_f32_16x16x32_bf16(false, a.v, false, b.v, (short)0, c, false, false);
}

template <int EPI>
__device__ __forceinline__ void epi_store(v8f acc, int z, int mbase, int n, int g,
                                          void* __restrict__ out, const float* __restrict__ aux) {
    if (EPI == EPI_Q) {                       // m=s, n=e -> Qphi[b][h][s][d] bf16
        const int h = n >> 6, d = n & 63;
        uint16_t* O = (uint16_t*)out;
        #pragma unroll
        for (int r = 0; r < 8; ++r) {
            const int s = mbase + 8 * g + r;
            O[(((size_t)z * NH + h) * SS + s) * HD + d] = f2bf(phi(acc[r]));
        }
    } else if (EPI == EPI_K || EPI == EPI_V) { // m=e, n=s -> Xt[b][h][d][s] bf16
        uint16_t* O = (uint16_t*)out;
        #pragma unroll
        for (int r = 0; r < 8; ++r) {
            const int e = mbase + 8 * g + r, h = e >> 6, d = e & 63;
            const float v = (EPI == EPI_K) ? phi(acc[r]) : acc[r];
            O[(((size_t)z * NH + h) * HD + d) * SS + n] = f2bf(v);
        }
    } else if (EPI == EPI_KVT) {              // m=d, n=e -> KVt[z][e][d] bf16, 16B packed
        uint16_t* O = (uint16_t*)out;
        Pack8 p;
        #pragma unroll
        for (int r = 0; r < 8; ++r) p.h[r] = f2bf(acc[r]);
        *(uint4*)(O + ((size_t)z * HD + n) * HD + mbase + 8 * g) = p.u;
    } else if (EPI == EPI_NUM) {              // m=s, n=e -> O[b][s][h*64+e] bf16, /(den+eps)
        const int b = z / NH, h = z % NH;
        uint16_t* O = (uint16_t*)out;
        #pragma unroll
        for (int r = 0; r < 8; ++r) {
            const int s = mbase + 8 * g + r;
            const float den = aux[(size_t)z * SS + s] + 1e-6f;
            O[((size_t)b * SS + s) * CC + h * HD + n] = f2bf(acc[r] / den);
        }
    } else {                                  // EPI_OUT: m=c, n=s -> y[b][c][s] f32 + bo[c]
        float* O = (float*)out;
        #pragma unroll
        for (int r = 0; r < 8; ++r) {
            const int c = mbase + 8 * g + r;
            O[((size_t)z * CC + c) * SS + n] = acc[r] + aux[c];
        }
    }
}

// ---- direct (L2-fed) GEMM for the small per-head problems: wave = 16x64 tile ----
template <int EPI>
__global__ void k_gemm(const uint16_t* __restrict__ A, const uint16_t* __restrict__ Bt,
                       size_t strideA, size_t strideB, int K,
                       void* __restrict__ out, const float* __restrict__ aux) {
    const int lane = threadIdx.x, g = lane >> 4, ln = lane & 15;
    const int z  = blockIdx.z;
    const int m0 = (blockIdx.y * blockDim.y + threadIdx.y) * 16;
    const int n0 = blockIdx.x * 64;

    const uint16_t* arow = A + (size_t)z * strideA + (size_t)(m0 + ln) * K;
    const uint16_t* Bb   = Bt + (size_t)z * strideB;

    v8f acc[4] = {v8f{}, v8f{}, v8f{}, v8f{}};
    for (int k0 = 0; k0 < K; k0 += 32) {
        ABFrag a;
        a.q[0] = *(const float4*)(arow + k0 + 16 * g);
        a.q[1] = *(const float4*)(arow + k0 + 16 * g + 8);
        #pragma unroll
        for (int tt = 0; tt < 4; ++tt) {
            const uint16_t* brow = Bb + (size_t)(n0 + tt * 16 + ln) * K + k0 + 16 * g;
            ABFrag b;
            b.q[0] = *(const float4*)(brow);
            b.q[1] = *(const float4*)(brow + 8);
            acc[tt] = wmma_bf16(a, b, acc[tt]);
        }
    }
    #pragma unroll
    for (int tt = 0; tt < 4; ++tt)
        epi_store<EPI>(acc[tt], z, m0, n0 + tt * 16 + ln, g, out, aux);
}

// ---- LDS-tiled GEMM for the big 512-class GEMMs ----
// Block = 256 threads (8 waves). Macro tile 128(M) x 128(N); wave (wm,wn) owns 32x64.
// K step = 32, double-buffered async global->LDS staging (4 x 16B per thread per step).
#define LDSP 40   // padded row stride in elements (80B, 16B-aligned)

template <int EPI>
__global__ void __launch_bounds__(256)
k_gemm_lds(const uint16_t* __restrict__ A, const uint16_t* __restrict__ Bt,
           size_t strideA, size_t strideB, int K,
           void* __restrict__ out, const float* __restrict__ aux) {
    __shared__ uint16_t Ash[2][128][LDSP];
    __shared__ uint16_t Bsh[2][128][LDSP];

    const int lane = threadIdx.x, g = lane >> 4, ln = lane & 15;
    const int w  = threadIdx.y;            // wave id 0..7
    const int wm = w & 3, wn = w >> 2;     // 4 x 2 wave grid over 128x128
    const int tid = w * 32 + lane;         // 0..255
    const int z  = blockIdx.z;
    const int m0 = blockIdx.y * 128;
    const int n0 = blockIdx.x * 128;

    const uint16_t* Ag = A + (size_t)z * strideA;
    const uint16_t* Bg = Bt + (size_t)z * strideB;

    // staging map: chunk c in [0,512): row=c>>2, part=c&3 (16B each); 2 chunks/thread/operand
    const int r0 = (tid) >> 2,        p0 = (tid) & 3;
    const int r1 = (tid + 256) >> 2,  p1 = (tid + 256) & 3;

    v8f acc[2][4] = {{v8f{}, v8f{}, v8f{}, v8f{}}, {v8f{}, v8f{}, v8f{}, v8f{}}};

    // preload k0 = 0 into buffer 0
    cp16_g2l(Ag + (size_t)(m0 + r0) * K + p0 * 8, &Ash[0][r0][p0 * 8]);
    cp16_g2l(Ag + (size_t)(m0 + r1) * K + p1 * 8, &Ash[0][r1][p1 * 8]);
    cp16_g2l(Bg + (size_t)(n0 + r0) * K + p0 * 8, &Bsh[0][r0][p0 * 8]);
    cp16_g2l(Bg + (size_t)(n0 + r1) * K + p1 * 8, &Bsh[0][r1][p1 * 8]);

    int buf = 0;
    for (int k0 = 0; k0 < K; k0 += 32) {
        wait_async_lds();
        __syncthreads();                    // tile (buf) resident for all waves
        if (k0 + 32 < K) {                  // prefetch next tile into buf^1
            const int kn = k0 + 32;
            cp16_g2l(Ag + (size_t)(m0 + r0) * K + kn + p0 * 8, &Ash[buf ^ 1][r0][p0 * 8]);
            cp16_g2l(Ag + (size_t)(m0 + r1) * K + kn + p1 * 8, &Ash[buf ^ 1][r1][p1 * 8]);
            cp16_g2l(Bg + (size_t)(n0 + r0) * K + kn + p0 * 8, &Bsh[buf ^ 1][r0][p0 * 8]);
            cp16_g2l(Bg + (size_t)(n0 + r1) * K + kn + p1 * 8, &Bsh[buf ^ 1][r1][p1 * 8]);
        }
        // compute from LDS (ds_load_b128 fragments)
        ABFrag af[2];
        #pragma unroll
        for (int ms = 0; ms < 2; ++ms) {
            const uint16_t* ap = &Ash[buf][wm * 32 + ms * 16 + ln][0];
            af[ms].q[0] = *(const float4*)(ap + 16 * g);
            af[ms].q[1] = *(const float4*)(ap + 16 * g + 8);
        }
        #pragma unroll
        for (int ns = 0; ns < 4; ++ns) {
            const uint16_t* bp = &Bsh[buf][wn * 64 + ns * 16 + ln][16 * g];
            ABFrag b;
            b.q[0] = *(const float4*)(bp);
            b.q[1] = *(const float4*)(bp + 8);
            #pragma unroll
            for (int ms = 0; ms < 2; ++ms)
                acc[ms][ns] = wmma_bf16(af[ms], b, acc[ms][ns]);
        }
        buf ^= 1;
    }

    #pragma unroll
    for (int ms = 0; ms < 2; ++ms)
        #pragma unroll
        for (int ns = 0; ns < 4; ++ns)
            epi_store<EPI>(acc[ms][ns], z, m0 + wm * 32 + ms * 16,
                           n0 + wn * 64 + ns * 16 + ln, g, out, aux);
}

// ---------------- launch ----------------

extern "C" void kernel_launch(void* const* d_in, const int* in_sizes, int n_in,
                              void* d_out, int out_size, void* d_ws, size_t ws_size,
                              hipStream_t stream) {
    (void)in_sizes; (void)n_in; (void)out_size; (void)ws_size;
    const float* x  = (const float*)d_in[0];
    const float* Wq = (const float*)d_in[1];
    const float* Wk = (const float*)d_in[2];
    const float* Wv = (const float*)d_in[3];
    const float* Wo = (const float*)d_in[4];
    const float* bo = (const float*)d_in[5];

    // workspace carve (~164 MB total)
    char* w = (char*)d_ws;
    auto carve = [&](size_t bytes) { char* p = w; w += (bytes + 255) & ~(size_t)255; return p; };
    uint16_t* t    = (uint16_t*)carve((size_t)BB * SS * CC * 2);      // t(B,S,C)
    uint16_t* Wq_b = (uint16_t*)carve((size_t)CC * CC * 2);
    uint16_t* Wk_b = (uint16_t*)carve((size_t)CC * CC * 2);
    uint16_t* Wv_b = (uint16_t*)carve((size_t)CC * CC * 2);
    uint16_t* Wo_b = (uint16_t*)carve((size_t)CC * CC * 2);
    uint16_t* Qb   = (uint16_t*)carve((size_t)BB * NH * SS * HD * 2); // phiQ (b,h,s,d)
    uint16_t* Ktb  = (uint16_t*)carve((size_t)BB * NH * HD * SS * 2); // phiK^T (b,h,d,s)
    uint16_t* Vtb  = (uint16_t*)carve((size_t)BB * NH * HD * SS * 2); // V^T (b,h,e,s)
    uint16_t* KVt  = (uint16_t*)carve((size_t)BB * NH * HD * HD * 2); // KV^T (z,e,d)
    uint16_t* Ob   = (uint16_t*)carve((size_t)BB * SS * CC * 2);      // O (b,s,c)
    float*    ksum = (float*)carve((size_t)BB * NH * HD * 4);
    float*    den  = (float*)carve((size_t)BB * NH * SS * 4);

    const size_t tSZ = (size_t)SS * CC;   // per-batch stride of t / Ob
    const size_t hSZ = (size_t)HD * SS;   // per-(b,h) stride of Kt/Vt/Q

    k_cvt4<<<(CC * CC + 255) / 256, 256, 0, stream>>>(Wq, Wk, Wv, Wo, Wq_b, Wk_b, Wv_b, Wo_b, CC * CC);
    k_transpose<<<dim3(SS / 32, CC / 32, BB), dim3(32, 8), 0, stream>>>(x, t);

    // Q = t @ Wq^T            (M=S, N=E, K=C)
    k_gemm_lds<EPI_Q  ><<<dim3(CC / 128, SS / 128, BB), dim3(32, 8), 0, stream>>>(t, Wq_b, tSZ, 0, CC, Qb, nullptr);
    // K^T = Wk x t-as-B^T     (M=E, N=S, K=C)
    k_gemm_lds<EPI_K  ><<<dim3(SS / 128, CC / 128, BB), dim3(32, 8), 0, stream>>>(Wk_b, t, 0, tSZ, CC, Ktb, nullptr);
    // V^T = Wv x t-as-B^T
    k_gemm_lds<EPI_V  ><<<dim3(SS / 128, CC / 128, BB), dim3(32, 8), 0, stream>>>(Wv_b, t, 0, tSZ, CC, Vtb, nullptr);

    k_ksum<<<(BB * NH * HD) / 8, 256, 0, stream>>>(Ktb, ksum);
    k_den <<<(BB * NH * SS) / 256, 256, 0, stream>>>(Qb, ksum, den);

    // KV = phiK^T @ V         (M=d, N=e, K=S), per (b,h)
    k_gemm<EPI_KVT><<<dim3(1, 1, BB * NH), dim3(32, 4), 0, stream>>>(Ktb, Vtb, hSZ, hSZ, SS, KVt, nullptr);
    // num = phiQ @ KV, /den   (M=S, N=e, K=d), per (b,h) -> O(b,s,c)
    k_gemm<EPI_NUM><<<dim3(1, SS / 64, BB * NH), dim3(32, 4), 0, stream>>>(Qb, KVt, hSZ, (size_t)(HD * HD), HD, Ob, den);
    // y = Wo x O-as-B^T + bo  (M=C, N=S, K=C) -> d_out (B,C,H,W)
    k_gemm<EPI_OUT><<<dim3(SS / 64, CC / 64, BB), dim3(32, 4), 0, stream>>>(Wo_b, Ob, 0, tSZ, CC, d_out, bo);
}
